// ChannelAttentionModule_1408749273876
// MI455X (gfx1250) — compile-verified, matrix operand order checked by reading, not run on hardware
//
#include <hip/hip_runtime.h>
#include <hip/hip_bf16.h>

// ---------------------------------------------------------------------------
// Channel attention (N=8, C=512, S=HW=4096), fp32 in/out.
//   E   = v v^T                      (bf16 WMMA, f32 acc)
//   att = softmax(max-E) = exp(minE - E)/sum  (row-wise)
//   out = gamma * (att v) + x        (bf16 WMMA, f32 acc)
// Working set fits in 192MB L2. Tiles staged with CDNA5 async global->LDS
// copies (ASYNCcnt) and double-buffered: one barrier per K-iteration.
// ---------------------------------------------------------------------------

typedef __attribute__((ext_vector_type(16))) __bf16 v16bf;
typedef __attribute__((ext_vector_type(8)))  float  v8f;
typedef __attribute__((ext_vector_type(4)))  __bf16 bf4;

#define CCH  512
#define SSP  4096
#define NBAT 8
#define BK   32
#define LDT  40   // BK + 8 pad: row stride 80B -> conflict-free b128 frag loads

// ---- CDNA5 async global->LDS (16B per lane), tracked by ASYNCcnt ----------
__device__ __forceinline__ void async_g2l_b128(unsigned lds_addr, const void* gaddr) {
    asm volatile("global_load_async_to_lds_b128 %0, %1, off"
                 :: "v"(lds_addr), "v"(gaddr) : "memory");
}
__device__ __forceinline__ void wait_async0() {
    asm volatile("s_wait_asynccnt 0" ::: "memory");
}
__device__ __forceinline__ unsigned lds_lo(const void* p) {
    return (unsigned)(unsigned long long)p;   // generic LDS ptr: addr[31:0] = LDS offset
}

// ---- fp32 -> bf16 conversion (vectorized) ---------------------------------
__global__ __launch_bounds__(256) void k_cvt(const float* __restrict__ x,
                                             __bf16* __restrict__ xb, long n) {
    long i = (long)blockIdx.x * blockDim.x + threadIdx.x;
    long stride = (long)gridDim.x * blockDim.x;
    const float4* x4 = (const float4*)x;
    long n4 = n >> 2;
    for (long v = i; v < n4; v += stride) {
        float4 f = x4[v];
        bf4 b;
        b.x = (__bf16)f.x; b.y = (__bf16)f.y;
        b.z = (__bf16)f.z; b.w = (__bf16)f.w;
        *(bf4*)(xb + (v << 2)) = b;
    }
}

// ---- per-lane WMMA fragment load from LDS [row][k] layout -----------------
// 16-bit A/B layout: lanes 0-15 hold row (lane&15), K 0..7 & 16..23;
// lanes 16-31 hold K 8..15 & 24..31  -> two contiguous b128 reads.
__device__ __forceinline__ v16bf frag_ld(const __bf16* base, int lane) {
    int r  = lane & 15;
    int kb = (lane >> 4) << 3;
    const __bf16* p = base + r * LDT + kb;
    v16bf f;
    ((uint4*)&f)[0] = *(const uint4*)(p);
    ((uint4*)&f)[1] = *(const uint4*)(p + 16);
    return f;
}

// ---- GEMM 1: E[b] = v v^T  (M=N=512, K=4096) ------------------------------
__global__ __launch_bounds__(256) void k_gram(const __bf16* __restrict__ xb,
                                              float* __restrict__ E) {
    __shared__ __bf16 lA[2][128 * LDT];
    __shared__ __bf16 lB[2][128 * LDT];
    const int b     = blockIdx.z;
    const int mBase = blockIdx.y * 128;
    const int nBase = blockIdx.x * 128;
    const __bf16* vb = xb + (long)b * CCH * SSP;

    const int tid = threadIdx.x, lane = tid & 31, wid = tid >> 5;
    const int wm = (wid & 3) * 32;   // 4 waves along M
    const int wn = (wid >> 2) * 64;  // 2 waves along N

    // This thread's two 16B staging chunks (rows row0 and row0+64, koff ko0)
    const int row0 = tid >> 2;
    const int ko0  = (tid & 3) << 3;
    const __bf16* gA0 = vb + (long)(mBase + row0)      * SSP + ko0;
    const __bf16* gA1 = vb + (long)(mBase + row0 + 64) * SSP + ko0;
    const __bf16* gB0 = vb + (long)(nBase + row0)      * SSP + ko0;
    const __bf16* gB1 = vb + (long)(nBase + row0 + 64) * SSP + ko0;
    unsigned dA0[2], dA1[2], dB0[2], dB1[2];
    #pragma unroll
    for (int bb = 0; bb < 2; bb++) {
        dA0[bb] = lds_lo(&lA[bb][row0 * LDT + ko0]);
        dA1[bb] = lds_lo(&lA[bb][(row0 + 64) * LDT + ko0]);
        dB0[bb] = lds_lo(&lB[bb][row0 * LDT + ko0]);
        dB1[bb] = lds_lo(&lB[bb][(row0 + 64) * LDT + ko0]);
    }

    v8f acc[2][4];
    const v8f vzero = {0.f,0.f,0.f,0.f,0.f,0.f,0.f,0.f};
    #pragma unroll
    for (int i = 0; i < 2; i++)
        #pragma unroll
        for (int j = 0; j < 4; j++) acc[i][j] = vzero;

    // prologue: stage tile 0 into buffer 0
    async_g2l_b128(dA0[0], gA0);
    async_g2l_b128(dA1[0], gA1);
    async_g2l_b128(dB0[0], gB0);
    async_g2l_b128(dB1[0], gB1);

    int bb = 0;
    for (int k0 = 0; k0 < SSP; k0 += BK, bb ^= 1) {
        wait_async0();        // this wave's LDS writes for buf[bb] landed
        __syncthreads();      // everyone's landed
        if (k0 + BK < SSP) {  // stage next tile into the other buffer
            int kn = k0 + BK, nb = bb ^ 1;
            async_g2l_b128(dA0[nb], gA0 + kn);
            async_g2l_b128(dA1[nb], gA1 + kn);
            async_g2l_b128(dB0[nb], gB0 + kn);
            async_g2l_b128(dB1[nb], gB1 + kn);
        }

        v16bf af[2], bf[4];
        #pragma unroll
        for (int mt = 0; mt < 2; mt++) af[mt] = frag_ld(&lA[bb][(wm + mt * 16) * LDT], lane);
        #pragma unroll
        for (int nt = 0; nt < 4; nt++) bf[nt] = frag_ld(&lB[bb][(wn + nt * 16) * LDT], lane);

        #pragma unroll
        for (int mt = 0; mt < 2; mt++)
            #pragma unroll
            for (int nt = 0; nt < 4; nt++)
                acc[mt][nt] = __builtin_amdgcn_wmma_f32_16x16x32_bf16(
                    false, af[mt], false, bf[nt], (short)0, acc[mt][nt], false, false);
    }

    // C/D layout: vgpr i, lanes0-15 -> M=i, lanes16-31 -> M=i+8; N = lane&15
    float* Eb = E + (long)b * CCH * CCH;
    const int r = lane & 15, mo = (lane >> 4) * 8;
    #pragma unroll
    for (int mt = 0; mt < 2; mt++)
        #pragma unroll
        for (int nt = 0; nt < 4; nt++) {
            int col  = nBase + wn + nt * 16 + r;
            int rowb = mBase + wm + mt * 16 + mo;
            #pragma unroll
            for (int i = 0; i < 8; i++)
                Eb[(long)(rowb + i) * CCH + col] = acc[mt][nt][i];
        }
}

// ---- reversed-sign softmax: att = exp(min_e - e) / sum --------------------
__global__ __launch_bounds__(256) void k_soft(const float* __restrict__ E,
                                              __bf16* __restrict__ att) {
    long row = blockIdx.x;
    const float* e = E + row * CCH;
    const int tid = threadIdx.x, lane = tid & 31, wid = tid >> 5;
    __shared__ float red[8];
    __shared__ float bcast;

    float v0 = e[tid], v1 = e[tid + 256];
    float lmin = fminf(v0, v1);
    #pragma unroll
    for (int off = 16; off; off >>= 1)
        lmin = fminf(lmin, __shfl_xor(lmin, off, 32));
    if (lane == 0) red[wid] = lmin;
    __syncthreads();
    if (tid == 0) {
        float m = red[0];
        #pragma unroll
        for (int i = 1; i < 8; i++) m = fminf(m, red[i]);
        bcast = m;
    }
    __syncthreads();
    float mn = bcast;
    float f0 = __expf(mn - v0), f1 = __expf(mn - v1);
    float ls = f0 + f1;
    #pragma unroll
    for (int off = 16; off; off >>= 1) ls += __shfl_xor(ls, off, 32);
    __syncthreads();               // everyone has read bcast; red reusable
    if (lane == 0) red[wid] = ls;
    __syncthreads();
    if (tid == 0) {
        float s = 0.f;
        #pragma unroll
        for (int i = 0; i < 8; i++) s += red[i];
        bcast = 1.0f / s;
    }
    __syncthreads();
    float inv = bcast;
    att[row * CCH + tid]       = (__bf16)(f0 * inv);
    att[row * CCH + tid + 256] = (__bf16)(f1 * inv);
}

// ---- GEMM 2: out = gamma*(att v) + x  (M=512 c, N=4096 s, K=512 d) --------
__global__ __launch_bounds__(256) void k_out(const __bf16* __restrict__ att,
                                             const __bf16* __restrict__ xb,
                                             const float* __restrict__ x,
                                             const float* __restrict__ gamma,
                                             float* __restrict__ out) {
    __shared__ __bf16 lA[2][128 * LDT];
    __shared__ __bf16 lB[2][128 * LDT];
    const int b     = blockIdx.z;
    const int mBase = blockIdx.y * 128;  // channel rows
    const int nBase = blockIdx.x * 128;  // spatial cols
    const __bf16* A = att + (long)b * CCH * CCH;
    const __bf16* V = xb  + (long)b * CCH * SSP;

    const int tid = threadIdx.x, lane = tid & 31, wid = tid >> 5;
    const int wm = (wid & 3) * 32;
    const int wn = (wid >> 2) * 64;

    // A staging (async): rows row0, row0+64 with k-offset ko0
    const int row0 = tid >> 2;
    const int ko0  = (tid & 3) << 3;
    const __bf16* gA0 = A + (long)(mBase + row0)      * CCH + ko0;
    const __bf16* gA1 = A + (long)(mBase + row0 + 64) * CCH + ko0;
    unsigned dA0[2], dA1[2];
    #pragma unroll
    for (int q = 0; q < 2; q++) {
        dA0[q] = lds_lo(&lA[q][row0 * LDT + ko0]);
        dA1[q] = lds_lo(&lA[q][(row0 + 64) * LDT + ko0]);
    }
    // B staging (transpose through VGPRs): v[d][s] -> lB[s][d]
    const int d0 = tid >> 4;            // 0..15 (chunk tid), +16 for chunk tid+256
    const int so0 = (tid & 15) << 3;    // 0..120

    v8f acc[2][4];
    const v8f vzero = {0.f,0.f,0.f,0.f,0.f,0.f,0.f,0.f};
    #pragma unroll
    for (int i = 0; i < 2; i++)
        #pragma unroll
        for (int j = 0; j < 4; j++) acc[i][j] = vzero;

    // prologue: stage tile 0 into buffer 0
    async_g2l_b128(dA0[0], gA0);
    async_g2l_b128(dA1[0], gA1);
    {
        uint4 v0 = *(const uint4*)(V + (long)d0 * SSP + nBase + so0);
        uint4 v1 = *(const uint4*)(V + (long)(d0 + 16) * SSP + nBase + so0);
        const __bf16* p0 = (const __bf16*)&v0;
        const __bf16* p1 = (const __bf16*)&v1;
        #pragma unroll
        for (int i = 0; i < 8; i++) {
            lB[0][(so0 + i) * LDT + d0]      = p0[i];
            lB[0][(so0 + i) * LDT + d0 + 16] = p1[i];
        }
    }

    int bb = 0;
    for (int k0 = 0; k0 < CCH; k0 += BK, bb ^= 1) {
        wait_async0();
        __syncthreads();
        if (k0 + BK < CCH) {
            int kn = k0 + BK, nb = bb ^ 1;
            async_g2l_b128(dA0[nb], gA0 + kn);
            async_g2l_b128(dA1[nb], gA1 + kn);
            uint4 v0 = *(const uint4*)(V + (long)(kn + d0) * SSP + nBase + so0);
            uint4 v1 = *(const uint4*)(V + (long)(kn + d0 + 16) * SSP + nBase + so0);
            const __bf16* p0 = (const __bf16*)&v0;
            const __bf16* p1 = (const __bf16*)&v1;
            #pragma unroll
            for (int i = 0; i < 8; i++) {
                lB[nb][(so0 + i) * LDT + d0]      = p0[i];
                lB[nb][(so0 + i) * LDT + d0 + 16] = p1[i];
            }
        }

        v16bf af[2], bf[4];
        #pragma unroll
        for (int mt = 0; mt < 2; mt++) af[mt] = frag_ld(&lA[bb][(wm + mt * 16) * LDT], lane);
        #pragma unroll
        for (int nt = 0; nt < 4; nt++) bf[nt] = frag_ld(&lB[bb][(wn + nt * 16) * LDT], lane);

        #pragma unroll
        for (int mt = 0; mt < 2; mt++)
            #pragma unroll
            for (int nt = 0; nt < 4; nt++)
                acc[mt][nt] = __builtin_amdgcn_wmma_f32_16x16x32_bf16(
                    false, af[mt], false, bf[nt], (short)0, acc[mt][nt], false, false);
    }

    // Epilogue: out = gamma*acc + x
    const float g = gamma[0];
    const float* xsrc = x   + (long)b * CCH * SSP;
    float*       ob   = out + (long)b * CCH * SSP;
    const int r = lane & 15, mo = (lane >> 4) * 8;
    #pragma unroll
    for (int mt = 0; mt < 2; mt++)
        #pragma unroll
        for (int nt = 0; nt < 4; nt++) {
            int col  = nBase + wn + nt * 16 + r;
            int rowb = mBase + wm + mt * 16 + mo;
            #pragma unroll
            for (int i = 0; i < 8; i++) {
                long idx = (long)(rowb + i) * SSP + col;
                ob[idx] = g * acc[mt][nt][i] + xsrc[idx];
            }
        }
}

// ---------------------------------------------------------------------------
extern "C" void kernel_launch(void* const* d_in, const int* in_sizes, int n_in,
                              void* d_out, int out_size, void* d_ws, size_t ws_size,
                              hipStream_t stream) {
    const float* x     = (const float*)d_in[0];
    const float* gamma = (const float*)d_in[1];
    float* out = (float*)d_out;

    char* ws = (char*)d_ws;
    size_t xb_bytes  = (size_t)NBAT * CCH * SSP * sizeof(__bf16);  // 32 MB
    size_t e_bytes   = (size_t)NBAT * CCH * CCH * sizeof(float);   //  8 MB
    __bf16* xb  = (__bf16*)ws;
    float*  E   = (float*)(ws + xb_bytes);
    __bf16* att = (__bf16*)(ws + xb_bytes + e_bytes);              // +4 MB

    long n = (long)NBAT * CCH * SSP;
    hipLaunchKernelGGL(k_cvt, dim3(2048), dim3(256), 0, stream, x, xb, n);

    dim3 g1(CCH / 128, CCH / 128, NBAT);   // 4 x 4 x 8
    hipLaunchKernelGGL(k_gram, g1, dim3(256), 0, stream, xb, E);

    hipLaunchKernelGGL(k_soft, dim3(NBAT * CCH), dim3(256), 0, stream, E, att);

    dim3 g2(SSP / 128, CCH / 128, NBAT);   // 32 x 4 x 8
    hipLaunchKernelGGL(k_out, g2, dim3(256), 0, stream, att, xb, x, gamma, out);
}
